// GRU_60567628808782
// MI455X (gfx1250) — matrix operand order repaired and checked by reading, not compiled, not run on hardware
//
#include <hip/hip_runtime.h>
#include <hip/hip_bf16.h>

// ---------------------------------------------------------------------------
// Problem constants (match the reference)
// ---------------------------------------------------------------------------
#define Bsz 16
#define Tn  1024
#define En  768
#define E3  2304          // 3*E
#define Vn  50257
#define KT  24            // E/32  k-tiles per GEMM
#define NT  144           // 3E/16 n-tiles per GEMM
#define NWG 16            // workgroups in the persistent GRU kernel
#define TPB 256           // threads per block
#define WPB 8             // waves per block (wave32)
#define NWAVE (NWG*WPB)   // 128 waves total
#define MATSZ ((size_t)E3*En)     // elements per packed weight matrix
#define FRAG  ((size_t)NT*32*8)   // f32 elements of one 16x2304 fragment buffer

typedef __attribute__((ext_vector_type(16))) __bf16 v16bf;
typedef __attribute__((ext_vector_type(8)))  __bf16 v8bf;
typedef __attribute__((ext_vector_type(8)))  float  v8f;

union AFrag { v16bf v; v8bf h[2]; };

// ---------------------------------------------------------------------------
// Fragment-layout indexing: C/D 16x16 f32 tile, VGPR r (0..7), lane layout per
// CDNA5 ISA: lanes 0-15 hold M=r, lanes 16-31 hold M=8+r; N = lane%16.
// Buffers store tiles as [tile][lane][r] (32B contiguous per lane).
// ---------------------------------------------------------------------------
__device__ __forceinline__ float frag_read(const float* __restrict__ buf, int b, int n) {
    int tile = n >> 4;
    int lane = (n & 15) + ((b & 8) << 1);   // +16 if b>=8
    int r    = b & 7;
    return buf[(size_t)tile * 256 + lane * 8 + r];
}

// ---------------------------------------------------------------------------
// Grid-wide barrier (monotone generation, arrive-release / spin-acquire)
// ---------------------------------------------------------------------------
__device__ __forceinline__ void grid_barrier(unsigned* bar, unsigned gen) {
    __syncthreads();
    __threadfence();
    if (threadIdx.x == 0) {
        __hip_atomic_fetch_add(bar, 1u, __ATOMIC_RELEASE, __HIP_MEMORY_SCOPE_AGENT);
        unsigned target = gen * NWG;
        while (__hip_atomic_load(bar, __ATOMIC_ACQUIRE, __HIP_MEMORY_SCOPE_AGENT) < target)
            __builtin_amdgcn_s_sleep(1);
    }
    __syncthreads();
    __threadfence();
}

// ---------------------------------------------------------------------------
// One 16x16 output tile of a (16x768)x(768x2304) GEMM.
// A is a single-address-space pointer at each inline site (global for x_t,
// LDS for h) so the loads lower to global_load_b128 / ds_load_b128, not flat.
// B is pre-packed in WMMA B-fragment layout (32B contiguous per lane).
// ---------------------------------------------------------------------------
__device__ __forceinline__ void gemm_tile(
    const __bf16* __restrict__ A,   // 16x768 bf16 row-major
    const __bf16* __restrict__ Wt,  // packed weight tiles for this n-tile
    float* __restrict__ out, int nt, int row, int half, int lane)
{
    v8f acc = {0.f, 0.f, 0.f, 0.f, 0.f, 0.f, 0.f, 0.f};
    const __bf16* arow = A + row * En + half * 8;
    #pragma unroll 4
    for (int kt = 0; kt < KT; ++kt) {
        AFrag af;
        af.h[0] = *(const v8bf*)(arow + kt * 32);        // K = half*8 .. +7
        af.h[1] = *(const v8bf*)(arow + kt * 32 + 16);   // K = 16+half*8 .. +7
        v16bf bf = *(const v16bf*)(Wt + (size_t)kt * 512 + lane * 16);
        if (kt + 1 < KT)
            __builtin_prefetch(Wt + (size_t)(kt + 1) * 512 + lane * 16, 0, 1);
        acc = __builtin_amdgcn_wmma_f32_16x16x32_bf16(
            false, af.v, false, bf, (short)0, acc, false, false);
    }
    *(v8f*)(out + (size_t)(nt * 32 + lane) * 8) = acc;   // fragment-layout store
}

// Two GEMMs of one phase, 288 tiles interleaved over 128 waves. `wave` is an
// SGPR value (via readfirstlane), so both the loop and the gemm-select branch
// are scalar: EXEC stays all-ones around every WMMA as the ISA requires.
__device__ __forceinline__ void gemm_phase(
    const __bf16* __restrict__ A0, const __bf16* __restrict__ A1,
    const __bf16* __restrict__ W0, const __bf16* __restrict__ W1,
    float* __restrict__ out0, float* __restrict__ out1,
    int wave, int lane)
{
    const int row  = lane & 15;
    const int half = lane >> 4;
    for (int tile = wave; tile < 2 * NT; tile += NWAVE) {
        if (tile < NT) {
            gemm_tile(A0, W0 + (size_t)tile * (KT * 512), out0, tile, row, half, lane);
        } else {
            const int nt = tile - NT;
            gemm_tile(A1, W1 + (size_t)nt * (KT * 512), out1, nt, row, half, lane);
        }
    }
}

// ---------------------------------------------------------------------------
// GRU gate combine (redundantly executed by every block into its own LDS h)
// ---------------------------------------------------------------------------
__device__ __forceinline__ void combine(
    const float* __restrict__ gi, const float* __restrict__ gh,
    const float* __restrict__ bi, const float* __restrict__ bh,
    __bf16* hb, float* hout, int tid)
{
    for (int i = tid; i < Bsz * En; i += TPB) {
        int b = i / En, e = i - b * En;
        float ir = frag_read(gi, b, e)          + bi[e];
        float iz = frag_read(gi, b, e + En)     + bi[e + En];
        float ig = frag_read(gi, b, e + 2 * En) + bi[e + 2 * En];
        float hr = frag_read(gh, b, e)          + bh[e];
        float hz = frag_read(gh, b, e + En)     + bh[e + En];
        float hg = frag_read(gh, b, e + 2 * En) + bh[e + 2 * En];
        float r = 1.f / (1.f + __expf(-(ir + hr)));
        float z = 1.f / (1.f + __expf(-(iz + hz)));
        float n = tanhf(ig + r * hg);
        float hold = (float)hb[i];
        float hnew = (1.f - z) * n + z * hold;
        hb[i] = (__bf16)hnew;
        if (hout) hout[i] = hnew;
    }
}

// ---------------------------------------------------------------------------
// Kernels
// ---------------------------------------------------------------------------
__global__ void init_ws_kernel(unsigned* bar) {
    if (threadIdx.x == 0 && blockIdx.x == 0) *bar = 0u;
}

// Gather token embeddings -> bf16 [T][B][E]
__global__ void embed_kernel(const int* __restrict__ idx,
                             const float* __restrict__ wte,
                             __bf16* __restrict__ Xb) {
    long long i = (long long)blockIdx.x * blockDim.x + threadIdx.x;
    if (i >= (long long)Tn * Bsz * En) return;
    int e = (int)(i % En);
    long long bt = i / En;
    int b = (int)(bt % Bsz);
    int t = (int)(bt / Bsz);
    int tok = idx[b * Tn + t];
    Xb[i] = (__bf16)wte[(size_t)tok * En + e];
}

// Repack Wih0/Whh0/Wih1/Whh1 (f32 [3E][E], gi = x @ W^T) into WMMA B-fragment
// tile layout: per matrix, tileIdx = nt*KT + kt, 512 bf16 per tile laid out
// [lane][j] with  n = (tileIdx/KT)*16 + lane%16,  k = (tileIdx%KT)*32 + (lane/16)*8 + (j%8) + (j/8)*16.
__global__ void pack_w_kernel(const float* __restrict__ Wih,
                              const float* __restrict__ Whh,
                              __bf16* __restrict__ packW) {
    long long i = (long long)blockIdx.x * blockDim.x + threadIdx.x;
    if (i >= (long long)4 * MATSZ) return;
    int mat = (int)(i / (long long)MATSZ);
    long long p = i % (long long)MATSZ;
    int tileIdx = (int)(p >> 9);
    int lane    = (int)((p >> 4) & 31);
    int j       = (int)(p & 15);
    int n = (tileIdx / KT) * 16 + (lane & 15);
    int k = (tileIdx % KT) * 32 + (lane >> 4) * 8 + (j & 7) + ((j >> 3) << 4);
    const float* src = (mat == 0) ? Wih
                     : (mat == 1) ? Whh
                     : (mat == 2) ? Wih + MATSZ
                                  : Whh + MATSZ;
    packW[i] = (__bf16)src[(size_t)n * En + k];
}

// Persistent sequential GRU: 16 blocks x 256 threads, 2 grid barriers / step.
__global__ void __launch_bounds__(TPB)
gru_seq_kernel(const __bf16* __restrict__ packW,
               const __bf16* __restrict__ Xb,
               float* __restrict__ gbuf,
               const float* __restrict__ bih,
               const float* __restrict__ bhh,
               float* __restrict__ hfin,
               unsigned* __restrict__ bar) {
    __shared__ __bf16 h0b[Bsz * En];
    __shared__ __bf16 h1b[Bsz * En];
    const int tid  = threadIdx.x;
    // SGPR wave index: tid>>5 is identical across the wave, so readfirstlane is
    // exact and makes all tile-scheduling control flow scalar (full EXEC at
    // every WMMA).
    const int waveInBlk = __builtin_amdgcn_readfirstlane(tid >> 5);
    const int wave = blockIdx.x * WPB + waveInBlk;
    const int lane = tid & 31;

    for (int i = tid; i < Bsz * En; i += TPB) {
        h0b[i] = (__bf16)0.f;
        h1b[i] = (__bf16)0.f;
    }
    __syncthreads();

    const __bf16* W_ih0 = packW;
    const __bf16* W_hh0 = packW + MATSZ;
    const __bf16* W_ih1 = packW + 2 * MATSZ;
    const __bf16* W_hh1 = packW + 3 * MATSZ;
    float* gi0 = gbuf;
    float* gh0 = gbuf + FRAG;
    float* gi1 = gbuf + 2 * FRAG;
    float* gh1 = gbuf + 3 * FRAG;

    unsigned bc = 0;
    for (int t = 0; t < Tn; ++t) {
        const __bf16* xt = Xb + (size_t)t * Bsz * En;

        // layer 0: gi0 = x_t @ Wih0^T (A global) ; gh0 = h0 @ Whh0^T (A in LDS)
        gemm_phase(xt, h0b, W_ih0, W_hh0, gi0, gh0, wave, lane);
        grid_barrier(bar, ++bc);
        combine(gi0, gh0, bih, bhh, h0b, nullptr, tid);
        __syncthreads();

        // layer 1: gi1 = h0_new @ Wih1^T ; gh1 = h1 @ Whh1^T (both A in LDS)
        gemm_phase(h0b, h1b, W_ih1, W_hh1, gi1, gh1, wave, lane);
        grid_barrier(bar, ++bc);
        float* hout = (t == Tn - 1 && blockIdx.x == 0) ? hfin : nullptr;
        combine(gi1, gh1, bih + E3, bhh + E3, h1b, hout, tid);
        __syncthreads();
    }
}

// LayerNorm (no bias) over last hidden state, one block per batch row.
__global__ void ln_kernel(const float* __restrict__ hfin,
                          const float* __restrict__ g,
                          float* __restrict__ lnout) {
    __shared__ float red[TPB];
    const int b = blockIdx.x, tid = threadIdx.x;
    const float* h = hfin + (size_t)b * En;
    float s = 0.f;
    for (int e = tid; e < En; e += TPB) s += h[e];
    red[tid] = s; __syncthreads();
    for (int o = TPB / 2; o > 0; o >>= 1) { if (tid < o) red[tid] += red[tid + o]; __syncthreads(); }
    float mu = red[0] / (float)En;
    __syncthreads();
    float v = 0.f;
    for (int e = tid; e < En; e += TPB) { float d = h[e] - mu; v += d * d; }
    red[tid] = v; __syncthreads();
    for (int o = TPB / 2; o > 0; o >>= 1) { if (tid < o) red[tid] += red[tid + o]; __syncthreads(); }
    float rstd = rsqrtf(red[0] / (float)En + 1e-5f);
    for (int e = tid; e < En; e += TPB)
        lnout[(size_t)b * En + e] = (h[e] - mu) * rstd * g[e];
}

// logits[b][v] = ln[b] . wte[v]  (HBM-bound: streams wte once, ln in LDS)
__global__ void __launch_bounds__(TPB)
logits_kernel(const float* __restrict__ lnb,
              const float* __restrict__ wte,
              float* __restrict__ out) {
    __shared__ float lnS[Bsz * En];
    const int tid = threadIdx.x;
    for (int i = tid; i < Bsz * En; i += TPB) lnS[i] = lnb[i];
    __syncthreads();
    int v = blockIdx.x * TPB + tid;
    if (v >= Vn) return;
    const float* wrow = wte + (size_t)v * En;
    float acc[Bsz];
    #pragma unroll
    for (int b = 0; b < Bsz; ++b) acc[b] = 0.f;
    for (int e = 0; e < En; e += 4) {
        float4 w4 = *(const float4*)(wrow + e);
        #pragma unroll
        for (int b = 0; b < Bsz; ++b) {
            const float* lp = &lnS[b * En + e];
            acc[b] += w4.x * lp[0] + w4.y * lp[1] + w4.z * lp[2] + w4.w * lp[3];
        }
    }
    #pragma unroll
    for (int b = 0; b < Bsz; ++b) out[(size_t)b * Vn + v] = acc[b];
}

// ---------------------------------------------------------------------------
// Host launcher
// ---------------------------------------------------------------------------
static inline size_t align256(size_t x) { return (x + 255) & ~(size_t)255; }

extern "C" void kernel_launch(void* const* d_in, const int* in_sizes, int n_in,
                              void* d_out, int out_size, void* d_ws, size_t ws_size,
                              hipStream_t stream) {
    (void)in_sizes; (void)n_in; (void)out_size; (void)ws_size;
    const int*   idx = (const int*)d_in[0];
    const float* wte = (const float*)d_in[1];
    const float* Wih = (const float*)d_in[2];
    const float* bih = (const float*)d_in[3];
    const float* Whh = (const float*)d_in[4];
    const float* bhh = (const float*)d_in[5];
    const float* g   = (const float*)d_in[6];
    float* out = (float*)d_out;

    // workspace carve-out
    char* ws = (char*)d_ws;
    size_t off = 0;
    unsigned* bar = (unsigned*)(ws + off);            off = align256(off + 256);
    __bf16* packW = (__bf16*)(ws + off);              off = align256(off + 4 * MATSZ * sizeof(__bf16));
    __bf16* Xb    = (__bf16*)(ws + off);              off = align256(off + (size_t)Tn * Bsz * En * sizeof(__bf16));
    float*  gbuf  = (float*)(ws + off);               off = align256(off + 4 * FRAG * sizeof(float));
    float*  hfin  = (float*)(ws + off);               off = align256(off + (size_t)Bsz * En * sizeof(float));
    float*  lnb   = (float*)(ws + off);               off = align256(off + (size_t)Bsz * En * sizeof(float));

    init_ws_kernel<<<1, 64, 0, stream>>>(bar);

    long long nEmb = (long long)Tn * Bsz * En;
    embed_kernel<<<(unsigned)((nEmb + TPB - 1) / TPB), TPB, 0, stream>>>(idx, wte, Xb);

    long long nPack = (long long)4 * MATSZ;
    pack_w_kernel<<<(unsigned)((nPack + TPB - 1) / TPB), TPB, 0, stream>>>(Wih, Whh, packW);

    gru_seq_kernel<<<NWG, TPB, 0, stream>>>(packW, Xb, gbuf, bih, bhh, hfin, bar);

    ln_kernel<<<Bsz, TPB, 0, stream>>>(hfin, g, lnb);

    logits_kernel<<<(Vn + TPB - 1) / TPB, TPB, 0, stream>>>(lnb, wte, out);
}
